// FullNN_69372311765598
// MI455X (gfx1250) — matrix operand order repaired and checked by reading, not compiled, not run on hardware
//
#include <hip/hip_runtime.h>

typedef __attribute__((ext_vector_type(16))) __bf16 v16bf;
typedef __attribute__((ext_vector_type(8)))  __bf16 v8bf;
typedef __attribute__((ext_vector_type(8)))  float  v8f;

#define PDIM 128
#define HDIM 512
#define QE_C 100000
#define N_IMG 1000
#define Q_TOT 300000
#define MT 64             // atoms per block (two 16-row strips per wave)
#define HS_STRIDE 520     // 512 + 8 bf16 pad -> rows offset by 4 banks
#define XS_STRIDE 136     // 128 + 8 bf16 pad
#define GX_STRIDE 132     // 128 + 4 f32 pad
#define APB 2048
#define NBP ((Q_TOT + APB - 1) / APB)   // 147

// dynamic LDS layout (bytes)
#define OFF_HS 0                        // 64 x HS_STRIDE bf16 = 66560
#define OFF_GS 66560                    // 64 x HS_STRIDE bf16 = 66560 (gx aliases this)
#define OFF_SB 133120                   // b0,b1,w2 : 3 x 512 f32 = 6144
#define OFF_EB 139264                   // 4 x 64 f32 = 1024
#define OFF_XS 140288                   // 64 x XS_STRIDE bf16 = 17408
#define SMEM_BYTES 157696

static __device__ __forceinline__ v8f vzero8f() {
  v8f z;
#pragma unroll
  for (int i = 0; i < 8; ++i) z[i] = 0.0f;
  return z;
}

// A-matrix fragment (16x32 bf16): lane half 0 -> K 0..7 in v0-3, K16..23 in v4-7
static __device__ __forceinline__ v16bf frag_a(const __bf16* p, int half) {
  v8bf lo = *(const v8bf*)(p + half * 8);
  v8bf hi = *(const v8bf*)(p + 16 + half * 8);
  v16bf f;
#pragma unroll
  for (int i = 0; i < 8; ++i) { f[i] = lo[i]; f[i + 8] = hi[i]; }
  return f;
}

// B-matrix fragment (32x16 bf16): lane half 0 -> K 0..7 in v0-3, K16..23 in v4-7
static __device__ __forceinline__ v16bf frag_b(const __bf16* p, int half) {
  v8bf lo = *(const v8bf*)(p + half * 8);
  v8bf hi = *(const v8bf*)(p + 16 + half * 8);
  v16bf f;
#pragma unroll
  for (int i = 0; i < 8; ++i) { f[i] = lo[i]; f[i + 8] = hi[i]; }
  return f;
}

#define WMMA_BF16(a, b, c) \
  __builtin_amdgcn_wmma_f32_16x16x32_bf16(false, (a), false, (b), (short)0, (c), false, false)

// -------------------- utility kernels --------------------

__global__ void zero_kernel(float* p, int n) {
  int i = blockIdx.x * 256 + threadIdx.x;
  if (i < n) p[i] = 0.0f;
}

// Convert w0,w1 (fp32) to bf16 row-major + transposed copies.
__global__ void convw_kernel(const float* __restrict__ w0, const float* __restrict__ w1,
                             __bf16* __restrict__ W0b, __bf16* __restrict__ W0t,
                             __bf16* __restrict__ W1b, __bf16* __restrict__ W1t) {
  int i = blockIdx.x * 256 + threadIdx.x;
  if (i < HDIM * HDIM) {
    int r = i >> 9, c = i & (HDIM - 1);
    __bf16 v = (__bf16)w1[i];
    W1b[i] = v;
    W1t[c * HDIM + r] = v;
  } else {
    int j = i - HDIM * HDIM;
    if (j < PDIM * HDIM) {
      int r = j >> 9, c = j & (HDIM - 1);
      __bf16 v = (__bf16)w0[j];
      W0b[j] = v;
      W0t[c * PDIM + r] = v;
    }
  }
}

// -------------------- fused MLP fwd+bwd --------------------
__global__ __launch_bounds__(256) void mlp_kernel(
    const float* __restrict__ x, const int* __restrict__ idx,
    const float* __restrict__ b0, const float* __restrict__ b1,
    const float* __restrict__ w2, const float* __restrict__ b2,
    const __bf16* __restrict__ W0b, const __bf16* __restrict__ W0t,
    const __bf16* __restrict__ W1b, const __bf16* __restrict__ W1t,
    float* __restrict__ gOut, float* __restrict__ energy) {
  extern __shared__ char smem[];
  __bf16* hs = (__bf16*)(smem + OFF_HS);
  __bf16* gs = (__bf16*)(smem + OFF_GS);
  float* gx = (float*)(smem + OFF_GS);    // aliases gs (dead after GEMM3 K-loop)
  float* sb0 = (float*)(smem + OFF_SB);
  float* sb1 = sb0 + HDIM;
  float* sw2 = sb1 + HDIM;
  float* ebuf = (float*)(smem + OFF_EB);  // [4][64]
  __bf16* xs = (__bf16*)(smem + OFF_XS);  // 64 x XS_STRIDE

  const int tid = threadIdx.x;
  const int lane = tid & 31;
  const int wave = tid >> 5;
  const int wm = wave >> 2;   // 0..1 : owns strips wm and wm+2 (rows 16*(wm+2t))
  const int wn = wave & 3;    // 0..3 : 128-col strip
  const int m15 = lane & 15;
  const int half = lane >> 4;
  const int a0 = blockIdx.x * MT;

  for (int i = tid; i < HDIM; i += 256) { sb0[i] = b0[i]; sb1[i] = b1[i]; sw2[i] = w2[i]; }
  for (int i = tid; i < MT * PDIM; i += 256) {
    int r = i >> 7, c = i & (PDIM - 1);
    int ar = a0 + r; if (ar >= QE_C) ar = QE_C - 1;   // clamp tail (dup real row, no NaNs)
    xs[r * XS_STRIDE + c] = (__bf16)x[(size_t)ar * PDIM + c];
  }
  __syncthreads();

  v8f acc[2][8];
  int rowb[2];
#pragma unroll
  for (int t = 0; t < 2; ++t) rowb[t] = (wm + 2 * t) * 16 + half * 8;

  // ---- GEMM1: z0 = X @ W0, h0 = tanh(z0 + b0) -> hs ----
#pragma unroll
  for (int t = 0; t < 2; ++t)
#pragma unroll
    for (int s = 0; s < 8; ++s) acc[t][s] = vzero8f();
  {
    const __bf16* ar0 = xs + ((wm + 0) * 16 + m15) * XS_STRIDE;
    const __bf16* ar1 = xs + ((wm + 2) * 16 + m15) * XS_STRIDE;
#pragma unroll
    for (int k0 = 0; k0 < PDIM; k0 += 32) {
      v16bf a0f = frag_a(ar0 + k0, half);
      v16bf a1f = frag_a(ar1 + k0, half);
#pragma unroll
      for (int s = 0; s < 8; ++s) {
        int n = wn * 128 + s * 16 + m15;
        v16bf b = frag_b(W0t + (size_t)n * PDIM + k0, half);
        acc[0][s] = WMMA_BF16(a0f, b, acc[0][s]);
        acc[1][s] = WMMA_BF16(a1f, b, acc[1][s]);
      }
    }
  }
#pragma unroll
  for (int t = 0; t < 2; ++t)
#pragma unroll
    for (int s = 0; s < 8; ++s) {
      int col = wn * 128 + s * 16 + m15;
#pragma unroll
      for (int r = 0; r < 8; ++r) {
        float h0 = tanhf(acc[t][s][r] + sb0[col]);
        hs[(rowb[t] + r) * HS_STRIDE + col] = (__bf16)h0;
      }
    }
  __syncthreads();

  // ---- GEMM2: z1 = h0 @ W1, h1 = tanh(z1 + b1); energy + gz1 = w2*(1-h1^2) -> gs ----
#pragma unroll
  for (int t = 0; t < 2; ++t)
#pragma unroll
    for (int s = 0; s < 8; ++s) acc[t][s] = vzero8f();
  {
    const __bf16* ar0 = hs + ((wm + 0) * 16 + m15) * HS_STRIDE;
    const __bf16* ar1 = hs + ((wm + 2) * 16 + m15) * HS_STRIDE;
    for (int k0 = 0; k0 < HDIM; k0 += 32) {
      v16bf a0f = frag_a(ar0 + k0, half);
      v16bf a1f = frag_a(ar1 + k0, half);
#pragma unroll
      for (int s = 0; s < 8; ++s) {
        int n = wn * 128 + s * 16 + m15;
        v16bf b = frag_b(W1t + (size_t)n * HDIM + k0, half);
        acc[0][s] = WMMA_BF16(a0f, b, acc[0][s]);
        acc[1][s] = WMMA_BF16(a1f, b, acc[1][s]);
      }
    }
  }
  {
    float ep[2][8];
#pragma unroll
    for (int t = 0; t < 2; ++t)
#pragma unroll
      for (int r = 0; r < 8; ++r) ep[t][r] = 0.0f;
#pragma unroll
    for (int t = 0; t < 2; ++t)
#pragma unroll
      for (int s = 0; s < 8; ++s) {
        int col = wn * 128 + s * 16 + m15;
        float w2c = sw2[col];
#pragma unroll
        for (int r = 0; r < 8; ++r) {
          float h1 = tanhf(acc[t][s][r] + sb1[col]);
          ep[t][r] += h1 * w2c;
          float gz1 = w2c * (1.0f - h1 * h1);
          gs[(rowb[t] + r) * HS_STRIDE + col] = (__bf16)gz1;
        }
      }
#pragma unroll
    for (int off = 1; off < 16; off <<= 1)
#pragma unroll
      for (int t = 0; t < 2; ++t)
#pragma unroll
        for (int r = 0; r < 8; ++r) ep[t][r] += __shfl_xor(ep[t][r], off, 32);
    if (m15 == 0) {
#pragma unroll
      for (int t = 0; t < 2; ++t)
#pragma unroll
        for (int r = 0; r < 8; ++r) ebuf[wn * 64 + rowb[t] + r] = ep[t][r];
    }
  }
  __syncthreads();
  if (tid < MT && a0 + tid < QE_C) {
    float es = ebuf[tid] + ebuf[64 + tid] + ebuf[128 + tid] + ebuf[192 + tid] + b2[0];
    atomicAdd(&energy[idx[a0 + tid]], es);
  }

  // ---- GEMM3: gh0 = gz1 @ w1^T ; gz0 = gh0*(1-h0^2) -> hs (overwrite own region) ----
#pragma unroll
  for (int t = 0; t < 2; ++t)
#pragma unroll
    for (int s = 0; s < 8; ++s) acc[t][s] = vzero8f();
  {
    const __bf16* ar0 = gs + ((wm + 0) * 16 + m15) * HS_STRIDE;
    const __bf16* ar1 = gs + ((wm + 2) * 16 + m15) * HS_STRIDE;
    for (int k0 = 0; k0 < HDIM; k0 += 32) {
      v16bf a0f = frag_a(ar0 + k0, half);
      v16bf a1f = frag_a(ar1 + k0, half);
#pragma unroll
      for (int s = 0; s < 8; ++s) {
        int n = wn * 128 + s * 16 + m15;   // B[j,n] = w1[n,j] -> row n of w1
        v16bf b = frag_b(W1b + (size_t)n * HDIM + k0, half);
        acc[0][s] = WMMA_BF16(a0f, b, acc[0][s]);
        acc[1][s] = WMMA_BF16(a1f, b, acc[1][s]);
      }
    }
  }
#pragma unroll
  for (int t = 0; t < 2; ++t)
#pragma unroll
    for (int s = 0; s < 8; ++s) {
      int col = wn * 128 + s * 16 + m15;
#pragma unroll
      for (int r = 0; r < 8; ++r) {
        float h0 = (float)hs[(rowb[t] + r) * HS_STRIDE + col];
        float gz0 = acc[t][s][r] * (1.0f - h0 * h0);
        hs[(rowb[t] + r) * HS_STRIDE + col] = (__bf16)gz0;
      }
    }
  __syncthreads();   // gz0 complete; gs no longer read -> gx may reuse its space

  // ---- GEMM4: gx = gz0 @ w0^T  (K split across wn waves, reduce in LDS) ----
  for (int i = tid; i < MT * GX_STRIDE; i += 256) gx[i] = 0.0f;
  __syncthreads();
#pragma unroll
  for (int t = 0; t < 2; ++t)
#pragma unroll
    for (int s = 0; s < 8; ++s) acc[t][s] = vzero8f();
  {
    const __bf16* ar0 = hs + ((wm + 0) * 16 + m15) * HS_STRIDE + wn * 128;
    const __bf16* ar1 = hs + ((wm + 2) * 16 + m15) * HS_STRIDE + wn * 128;
#pragma unroll
    for (int k0 = 0; k0 < 128; k0 += 32) {
      v16bf a0f = frag_a(ar0 + k0, half);
      v16bf a1f = frag_a(ar1 + k0, half);
#pragma unroll
      for (int s = 0; s < 8; ++s) {
        int n = s * 16 + m15;              // B[k,p] = w0[p,k] -> row p of w0
        v16bf b = frag_b(W0b + (size_t)n * HDIM + wn * 128 + k0, half);
        acc[0][s] = WMMA_BF16(a0f, b, acc[0][s]);
        acc[1][s] = WMMA_BF16(a1f, b, acc[1][s]);
      }
    }
  }
#pragma unroll
  for (int t = 0; t < 2; ++t)
#pragma unroll
    for (int s = 0; s < 8; ++s) {
      int col = s * 16 + m15;
#pragma unroll
      for (int r = 0; r < 8; ++r) atomicAdd(&gx[(rowb[t] + r) * GX_STRIDE + col], acc[t][s][r]);
    }
  __syncthreads();
  for (int i = tid; i < MT * PDIM; i += 256) {
    int r = i >> 7, c = i & (PDIM - 1);
    if (a0 + r < QE_C) gOut[(size_t)(a0 + r) * PDIM + c] = gx[r * GX_STRIDE + c];
  }
}

// -------------------- stable counting sort of image indices --------------------

static __device__ __forceinline__ int atom_img(int a, const int* i0, const int* i1, const int* i2) {
  return a < QE_C ? i0[a] : (a < 2 * QE_C ? i1[a - QE_C] : i2[a - 2 * QE_C]);
}

__global__ void hist_kernel(const int* __restrict__ i0, const int* __restrict__ i1,
                            const int* __restrict__ i2, unsigned* __restrict__ hist) {
  __shared__ unsigned h[N_IMG];
  for (int i = threadIdx.x; i < N_IMG; i += 256) h[i] = 0;
  __syncthreads();
  int base = blockIdx.x * APB;
  for (int i = threadIdx.x; i < APB; i += 256) {
    int a = base + i;
    if (a < Q_TOT) atomicAdd(&h[atom_img(a, i0, i1, i2)], 1u);
  }
  __syncthreads();
  for (int i = threadIdx.x; i < N_IMG; i += 256) hist[blockIdx.x * N_IMG + i] = h[i];
}

__global__ void scan_kernel(const unsigned* __restrict__ hist, unsigned* __restrict__ blockStart) {
  __shared__ unsigned tot[N_IMG];
  __shared__ unsigned base[N_IMG];
  for (int i = threadIdx.x; i < N_IMG; i += 256) {
    unsigned s = 0;
    for (int b = 0; b < NBP; ++b) s += hist[b * N_IMG + i];
    tot[i] = s;
  }
  __syncthreads();
  if (threadIdx.x == 0) {
    unsigned run = 0;
    for (int i = 0; i < N_IMG; ++i) { base[i] = run; run += tot[i]; }
  }
  __syncthreads();
  for (int i = threadIdx.x; i < N_IMG; i += 256) {
    unsigned run = base[i];
    for (int b = 0; b < NBP; ++b) { blockStart[b * N_IMG + i] = run; run += hist[b * N_IMG + i]; }
  }
}

__global__ void place_kernel(const int* __restrict__ i0, const int* __restrict__ i1,
                             const int* __restrict__ i2, const unsigned* __restrict__ blockStart,
                             unsigned* __restrict__ sortedAtom) {
  __shared__ unsigned offs[N_IMG];
  int lane = threadIdx.x;  // 32 threads = 1 wave (lockstep -> stable)
  for (int i = lane; i < N_IMG; i += 32) offs[i] = blockStart[blockIdx.x * N_IMG + i];
  __syncthreads();
  int base = blockIdx.x * APB;
  for (int c = 0; c < APB; c += 32) {
    int a = base + c + lane;
    bool valid = a < Q_TOT;
    int img = valid ? atom_img(a, i0, i1, i2) : (-1 - lane);
    int rank = 0, cnt = 0;
    for (int j = 0; j < 32; ++j) {
      int v = __shfl(img, j, 32);
      if (v == img) { cnt++; if (j < lane) rank++; }
    }
    if (valid) sortedAtom[offs[img] + rank] = (unsigned)a;
    __syncthreads();
    if (valid && rank == cnt - 1) offs[img] += (unsigned)cnt;
    __syncthreads();
  }
}

__global__ void src_kernel(const int* __restrict__ rearange, const unsigned* __restrict__ sortedAtom,
                           unsigned* __restrict__ srcAtom) {
  int q = blockIdx.x * 256 + threadIdx.x;
  if (q < Q_TOT) srcAtom[q] = sortedAtom[rearange[q]];
}

// -------------------- sparse force scatter --------------------
__global__ void force_kernel(const float* __restrict__ vals, const int* __restrict__ rows,
                             const int* __restrict__ cols, const unsigned* __restrict__ srcAtom,
                             const float* __restrict__ g, float* __restrict__ force, int nnz) {
  int n = blockIdx.x * 256 + threadIdx.x;
  if (n < nnz) {
    int r = rows[n];
    unsigned a = srcAtom[r >> 7];
    float v = vals[n] * g[(size_t)a * PDIM + (r & (PDIM - 1))];
    atomicAdd(&force[cols[n]], -v);
  }
}

// -------------------- host --------------------
extern "C" void kernel_launch(void* const* d_in, const int* in_sizes, int n_in,
                              void* d_out, int out_size, void* d_ws, size_t ws_size,
                              hipStream_t stream) {
  (void)n_in; (void)out_size; (void)ws_size;
  const float* x[3]; const int* idx[3];
  const float *w0[3], *b0[3], *w1[3], *b1[3], *w2[3], *b2[3];
  for (int e = 0; e < 3; ++e) {
    int k = e * 8;
    x[e]  = (const float*)d_in[k + 0];
    idx[e]= (const int*)  d_in[k + 1];
    w0[e] = (const float*)d_in[k + 2];
    b0[e] = (const float*)d_in[k + 3];
    w1[e] = (const float*)d_in[k + 4];
    b1[e] = (const float*)d_in[k + 5];
    w2[e] = (const float*)d_in[k + 6];
    b2[e] = (const float*)d_in[k + 7];
  }
  const int* rearange = (const int*)d_in[24];
  const int* fp_rows  = (const int*)d_in[25];
  const int* fp_cols  = (const int*)d_in[26];
  const float* fp_vals= (const float*)d_in[27];
  const int nnz = in_sizes[25];

  float* out = (float*)d_out;        // [0,1000): energy ; [1000, 901000): force
  char* ws = (char*)d_ws;

  const size_t szG = (size_t)Q_TOT * PDIM * sizeof(float);       // 153.6 MB
  const size_t WB = ((size_t)PDIM * HDIM * 2 + (size_t)HDIM * HDIM * 2) * sizeof(__bf16);
  float* g = (float*)ws;
  const size_t offHist = szG + 3 * WB;
  const size_t szHist = (size_t)NBP * N_IMG * sizeof(unsigned);
  unsigned* hist       = (unsigned*)(ws + offHist);
  unsigned* blockStart = (unsigned*)(ws + offHist + szHist);
  unsigned* sortedAtom = (unsigned*)(ws + offHist + 2 * szHist);
  unsigned* srcAtom    = (unsigned*)(ws + offHist + 2 * szHist + (size_t)Q_TOT * 4);

  // 1) zero outputs
  zero_kernel<<<(N_IMG + 3 * Q_TOT + 255) / 256, 256, 0, stream>>>(out, N_IMG + 3 * Q_TOT);

  // 2) weight conversion + MLP fwd/bwd per element
  for (int e = 0; e < 3; ++e) {
    char* wb = ws + szG + (size_t)e * WB;
    __bf16* W0b = (__bf16*)(wb);
    __bf16* W0t = (__bf16*)(wb + (size_t)PDIM * HDIM * 2);
    __bf16* W1b = (__bf16*)(wb + (size_t)PDIM * HDIM * 4);
    __bf16* W1t = (__bf16*)(wb + (size_t)PDIM * HDIM * 4 + (size_t)HDIM * HDIM * 2);
    convw_kernel<<<(HDIM * HDIM + PDIM * HDIM + 255) / 256, 256, 0, stream>>>(
        w0[e], w1[e], W0b, W0t, W1b, W1t);
    mlp_kernel<<<(QE_C + MT - 1) / MT, 256, SMEM_BYTES, stream>>>(
        x[e], idx[e], b0[e], b1[e], w2[e], b2[e], W0b, W0t, W1b, W1t,
        g + (size_t)e * QE_C * PDIM, out);
  }

  // 3) stable sort-by-image permutation
  hist_kernel<<<NBP, 256, 0, stream>>>(idx[0], idx[1], idx[2], hist);
  scan_kernel<<<1, 256, 0, stream>>>(hist, blockStart);
  place_kernel<<<NBP, 32, 0, stream>>>(idx[0], idx[1], idx[2], blockStart, sortedAtom);
  src_kernel<<<(Q_TOT + 255) / 256, 256, 0, stream>>>(rearange, sortedAtom, srcAtom);

  // 4) sparse force accumulation
  force_kernel<<<(nnz + 255) / 256, 256, 0, stream>>>(
      fp_vals, fp_rows, fp_cols, srcAtom, g, out + N_IMG, nnz);
}
